// SymmetryAwareBottleneck_42932493091506
// MI455X (gfx1250) — matrix operand order repaired and checked by reading, not compiled, not run on hardware
//
#include <hip/hip_runtime.h>
#include <hip/hip_bf16.h>

typedef __attribute__((ext_vector_type(16))) _Float16 v16h;
typedef __attribute__((ext_vector_type(8)))  float    v8f;

#define WMMA_F32_F16(a,b,c) \
  __builtin_amdgcn_wmma_f32_16x16x32_f16(false,(a),false,(b),(short)0,(c),false,false)

union Cf { v8f  v; float e[8]; };

static constexpr int   Cch   = 1024;
static constexpr int   Bn    = 2;
static constexpr int   Dd    = 8, Hh = 16, Wfull = 16, Wside = 8;
static constexpr int   Sside = Dd*Hh*Wside;   // 1024
static constexpr int   Sfull = Dd*Hh*Wfull;   // 2048
static constexpr int   HDIM  = 128;
static constexpr float ASCALE = 0.08838834764831845f; // 128^-0.5
static constexpr int   GROUPS = 32, GCH = Cch/GROUPS; // 32
static constexpr float EPS   = 1e-5f;

// LDS row strides (halves): 16B-aligned, bank-rotating
static constexpr int LDT  = 72;   // 64-half rows (K_TILE=64 GEMM tiles, V/P)
static constexpr int LDQ  = 136;  // 128-half rows (Q/K attention tiles)
static constexpr int LDS_S = 68;  // 64x64 f32 logits tile

// ---- WMMA fragments: two 8-k runs at k0+8g, k0+16+8g => 2x ds_load_b128 ------------------------
__device__ __forceinline__ v16h frag_a(const _Float16* As, int lda, int m0, int k0, int lane){
  union { v16h v; uint4 q[2]; } a;
  const int m = m0 + (lane & 15);
  const int g = (lane >> 4) & 1;
  const _Float16* row = As + m*lda + k0 + 8*g;
  a.q[0] = *(const uint4*)(row);
  a.q[1] = *(const uint4*)(row + 16);
  return a.v;
}

__device__ __forceinline__ v16h frag_bk(const _Float16* Bt, int ldb, int k0, int n0, int lane){
  union { v16h v; uint4 q[2]; } b;
  const int n  = n0 + (lane & 15);
  const int kg = (lane >> 4) & 1;
  const _Float16* row = Bt + n*ldb + k0 + 8*kg;
  b.q[0] = *(const uint4*)(row);
  b.q[1] = *(const uint4*)(row + 16);
  return b.v;
}

__device__ __forceinline__ void cp8(const _Float16* __restrict__ g, _Float16* l){
  *(uint4*)l = *(const uint4*)g;
}
__device__ __forceinline__ void cvt8r(float4 a, float4 b, _Float16* l){
  union { _Float16 h[8]; uint4 u; } t;
  t.h[0]=(_Float16)a.x; t.h[1]=(_Float16)a.y; t.h[2]=(_Float16)a.z; t.h[3]=(_Float16)a.w;
  t.h[4]=(_Float16)b.x; t.h[5]=(_Float16)b.y; t.h[6]=(_Float16)b.z; t.h[7]=(_Float16)b.w;
  *(uint4*)l = t.u;
}

// ---- prepass: x -> f16 [b][sp][c];  conv weights -> f16 [side][tap][o][c] ---
__global__ void __launch_bounds__(256) prep_x_kernel(
    const float* __restrict__ x, _Float16* __restrict__ xT)
{
  int i = blockIdx.x*256 + threadIdx.x;
  int b  = i >> 21;
  int sp = (i >> 10) & 2047;
  int c  = i & 1023;
  xT[i] = (_Float16)x[(size_t)(b*Cch + c)*Sfull + sp];
}

__global__ void __launch_bounds__(256) prep_w_kernel(
    const float* __restrict__ wl, const float* __restrict__ wr,
    _Float16* __restrict__ wT)
{
  const int tap = blockIdx.y, side = blockIdx.z;
  const float* w = side ? wr : wl;
  int i = blockIdx.x*256 + threadIdx.x;
  int o = i >> 10, c = i & 1023;
  wT[(((size_t)side*27 + tap) << 20) + i] = (_Float16)w[(size_t)(o*Cch + c)*27 + tap];
}

// ---- Kernel 1: conv3d implicit GEMM, K_TILE=64, branchless reg-prefetch -----
__global__ void __launch_bounds__(256) conv3d_kernel(
    const _Float16* __restrict__ xT, const _Float16* __restrict__ wT,
    const float* __restrict__ bl, const float* __restrict__ br,
    float* __restrict__ convbuf)
{
  __shared__ __align__(16) _Float16 As [64*LDT];
  __shared__ __align__(16) _Float16 Bst[64*LDT];
  const int tid  = threadIdx.x;
  const int lane = tid & 31;
  const int wv   = tid >> 5;
  const int ntile = blockIdx.x, mtile = blockIdx.y;
  const int side  = blockIdx.z >> 1, b = blockIdx.z & 1;
  const float* bias = side ? br : bl;
  const int m0g = mtile*64, n0g = ntile*64;
  const int m0  = 16*(wv & 3);
  const int n0w = 32*(wv >> 2);
  const int row = tid >> 2;
  const int ch0 = (tid & 3)*8, ch1 = ch0 + 32;
  const int s  = n0g + row;
  const int sd = s >> 7, sh = (s >> 3) & 15, sw = s & 7;
  // per-thread bases (single base pointer each -> global_load promotion)
  const _Float16* wrowb = wT + (((size_t)side*27) << 20) + (((size_t)(m0g + row)) << 10);
  const _Float16* xb    = xT + (((size_t)b*Sfull) << 10);
  const int sbase = sd*256 + sh*16 + side*Wside + sw;  // full-W spatial index

  Cf c0, c1;
  #pragma unroll
  for (int i = 0; i < 8; ++i){ c0.e[i] = 0.f; c1.e[i] = 0.f; }

  uint4 ra0, ra1, rb0, rb1;
  auto fetch = [&](int it){
    const int tap = it >> 4, cbase = (it & 15) << 6;
    const int kd = tap/9 - 1, kh = (tap/3)%3 - 1, kw = tap%3 - 1;
    const _Float16* wb = wrowb + (((size_t)tap) << 20) + cbase;
    ra0 = *(const uint4*)(wb + ch0);
    ra1 = *(const uint4*)(wb + ch1);
    const int d = sd + kd, h = sh + kh, w2 = sw + kw;
    const bool ok = ((unsigned)d < 8u) & ((unsigned)h < 16u) & ((unsigned)w2 < 8u);
    // branchless: clamp offset to 0 when OOB, then mask the payload
    const size_t xoff = ok ? (((size_t)(sbase + kd*256 + kh*16 + kw)) << 10) + cbase
                           : (size_t)0;
    rb0 = *(const uint4*)(xb + xoff + ch0);
    rb1 = *(const uint4*)(xb + xoff + ch1);
    const unsigned msk = ok ? 0xFFFFFFFFu : 0u;
    rb0.x &= msk; rb0.y &= msk; rb0.z &= msk; rb0.w &= msk;
    rb1.x &= msk; rb1.y &= msk; rb1.z &= msk; rb1.w &= msk;
  };
  fetch(0);

  for (int it = 0; it < 27*16; ++it){
    *(uint4*)(As  + row*LDT + ch0) = ra0;
    *(uint4*)(As  + row*LDT + ch1) = ra1;
    *(uint4*)(Bst + row*LDT + ch0) = rb0;
    *(uint4*)(Bst + row*LDT + ch1) = rb1;
    __syncthreads();
    if (it + 1 < 27*16) fetch(it + 1);   // loads in flight over compute
    {
      v16h a0 = frag_a(As, LDT, m0, 0, lane);
      v16h bb;
      bb = frag_bk(Bst, LDT, 0, n0w,      lane); c0.v = WMMA_F32_F16(a0, bb, c0.v);
      bb = frag_bk(Bst, LDT, 0, n0w + 16, lane); c1.v = WMMA_F32_F16(a0, bb, c1.v);
    }
    {
      v16h a1 = frag_a(As, LDT, m0, 32, lane);
      v16h bb;
      bb = frag_bk(Bst, LDT, 32, n0w,      lane); c0.v = WMMA_F32_F16(a1, bb, c0.v);
      bb = frag_bk(Bst, LDT, 32, n0w + 16, lane); c1.v = WMMA_F32_F16(a1, bb, c1.v);
    }
    __syncthreads();
  }
  const int g   = (lane >> 4) & 1;
  const int nc0 = n0g + n0w + (lane & 15);
  float* outb = convbuf + (size_t)((side*Bn + b)*Cch)*Sside;
  #pragma unroll
  for (int r = 0; r < 8; ++r){
    int o = m0g + m0 + r + 8*g;
    float bv = bias[o];
    outb[(size_t)o*Sside + nc0]      = c0.e[r] + bv;
    outb[(size_t)o*Sside + nc0 + 16] = c1.e[r] + bv;
  }
}

// ---- Kernel 2: per-side GN + ReLU -> f16 transposed [s][c], right W-flipped -
__global__ void __launch_bounds__(256) gn_side_kernel(
    const float* __restrict__ convbuf,
    const float* __restrict__ gl, const float* __restrict__ bl,
    const float* __restrict__ gr, const float* __restrict__ br,
    _Float16* __restrict__ gnT)
{
  __shared__ float red[256], red2[256];
  __shared__ float s_mu, s_rs;
  const int tid  = threadIdx.x;
  const int grp  = blockIdx.x & 31;
  const int b    = (blockIdx.x >> 5) & 1;
  const int side = blockIdx.x >> 6;
  const float* gamma = side ? gr : gl;
  const float* beta  = side ? br : bl;
  const float* src = convbuf + (size_t)((side*Bn + b)*Cch + grp*GCH)*Sside;
  const int N = GCH * Sside;
  float s = 0.f, ss = 0.f;
  for (int i = tid; i < N; i += 256){ float v = src[i]; s += v; ss += v*v; }
  red[tid] = s; red2[tid] = ss;
  __syncthreads();
  for (int off = 128; off > 0; off >>= 1){
    if (tid < off){ red[tid] += red[tid+off]; red2[tid] += red2[tid+off]; }
    __syncthreads();
  }
  if (tid == 0){
    float mu = red[0]/(float)N;
    s_mu = mu;
    s_rs = rsqrtf(red2[0]/(float)N - mu*mu + EPS);
  }
  __syncthreads();
  const float mu = s_mu, rs = s_rs;
  _Float16* dst = gnT + ((size_t)(side*Bn + b) << 20);
  for (int i = tid; i < N; i += 256){
    int cc = i >> 10, sp = i & 1023;
    int ch = grp*GCH + cc;
    float v = (src[i] - mu)*rs*gamma[ch] + beta[ch];
    v = fmaxf(v, 0.f);
    int spo = side ? (sp ^ 7) : sp;
    dst[((size_t)spo << 10) + ch] = (_Float16)v;
  }
}

// ---- Kernel 3: asym map ------------------------------------------------------
__global__ void __launch_bounds__(256) asym_kernel(
    const _Float16* __restrict__ gnT, float* __restrict__ out_asym)
{
  int i = blockIdx.x*256 + threadIdx.x;
  int b = i >> 10, s = i & 1023;
  const _Float16* fl = gnT + ((size_t)b        << 20) + ((size_t)s << 10);
  const _Float16* fr = gnT + ((size_t)(Bn + b) << 20) + ((size_t)s << 10);
  float acc = 0.f;
  for (int c = 0; c < Cch; c += 8){
    union { uint4 u; _Float16 h[8]; } A, B;
    A.u = *(const uint4*)(fl + c);
    B.u = *(const uint4*)(fr + c);
    #pragma unroll
    for (int j = 0; j < 8; ++j) acc += fabsf((float)A.h[j] - (float)B.h[j]);
  }
  out_asym[i] = acc * (1.f/(float)Cch);
}

// ---- Kernel 4: six projections, K_TILE=64, register-prefetch ----------------
__global__ void __launch_bounds__(256) proj_kernel(
    const _Float16* __restrict__ gnT,
    const float* qlw, const float* qlb, const float* krw, const float* krb,
    const float* vrw, const float* vrb, const float* qrw, const float* qrb,
    const float* klw, const float* klb, const float* vlw, const float* vlb,
    _Float16* __restrict__ projT)
{
  __shared__ __align__(16) _Float16 As [64*LDT];
  __shared__ __align__(16) _Float16 Bst[64*LDT];
  const int tid = threadIdx.x, lane = tid & 31, wv = tid >> 5;
  const int ntile = blockIdx.x, mtile = blockIdx.y;
  const int p = blockIdx.z >> 1, b = blockIdx.z & 1;
  const float* W; const float* bias; int srcSide;
  switch (p){
    case 0:  W = qlw; bias = qlb; srcSide = 0; break;
    case 1:  W = krw; bias = krb; srcSide = 1; break;
    case 2:  W = vrw; bias = vrb; srcSide = 1; break;
    case 3:  W = qrw; bias = qrb; srcSide = 1; break;
    case 4:  W = klw; bias = klb; srcSide = 0; break;
    default: W = vlw; bias = vlb; srcSide = 0; break;
  }
  const bool  transp = (p == 0) | (p == 1) | (p == 3) | (p == 4);
  const float scl    = ((p == 0) | (p == 3)) ? ASCALE : 1.f;
  const _Float16* src = gnT + ((size_t)(srcSide*Bn + b) << 20);
  const int m0g = mtile*64, n0g = ntile*64;
  const int m0 = 16*(wv & 3), n0w = 32*(wv >> 2);
  const int row = tid >> 2;
  const int ch0 = (tid & 3)*8, ch1 = ch0 + 32;
  const float*    wrow = W   + (size_t)(m0g + row)*Cch;
  const _Float16* brow = src + (((size_t)(n0g + row)) << 10);
  Cf c0, c1;
  #pragma unroll
  for (int i = 0; i < 8; ++i){ c0.e[i] = 0.f; c1.e[i] = 0.f; }

  float4 wa0, wa1, wa2, wa3; uint4 rb0, rb1;
  auto fetch = [&](int k0){
    wa0 = *(const float4*)(wrow + k0 + ch0);
    wa1 = *(const float4*)(wrow + k0 + ch0 + 4);
    wa2 = *(const float4*)(wrow + k0 + ch1);
    wa3 = *(const float4*)(wrow + k0 + ch1 + 4);
    rb0 = *(const uint4*)(brow + k0 + ch0);
    rb1 = *(const uint4*)(brow + k0 + ch1);
  };
  fetch(0);
  for (int it = 0; it < 16; ++it){
    cvt8r(wa0, wa1, As + row*LDT + ch0);
    cvt8r(wa2, wa3, As + row*LDT + ch1);
    *(uint4*)(Bst + row*LDT + ch0) = rb0;
    *(uint4*)(Bst + row*LDT + ch1) = rb1;
    __syncthreads();
    if (it + 1 < 16) fetch((it + 1)*64);
    {
      v16h a0 = frag_a(As, LDT, m0, 0, lane);
      v16h bb;
      bb = frag_bk(Bst, LDT, 0, n0w,      lane); c0.v = WMMA_F32_F16(a0, bb, c0.v);
      bb = frag_bk(Bst, LDT, 0, n0w + 16, lane); c1.v = WMMA_F32_F16(a0, bb, c1.v);
    }
    {
      v16h a1 = frag_a(As, LDT, m0, 32, lane);
      v16h bb;
      bb = frag_bk(Bst, LDT, 32, n0w,      lane); c0.v = WMMA_F32_F16(a1, bb, c0.v);
      bb = frag_bk(Bst, LDT, 32, n0w + 16, lane); c1.v = WMMA_F32_F16(a1, bb, c1.v);
    }
    __syncthreads();
  }
  const int g = (lane >> 4) & 1;
  const int nc0 = n0g + n0w + (lane & 15);
  _Float16* outb = projT + ((size_t)(p*Bn + b) << 20);
  #pragma unroll
  for (int r = 0; r < 8; ++r){
    int o = m0g + m0 + r + 8*g;
    float bv = bias[o];
    float v0 = (c0.e[r] + bv)*scl;
    float v1 = (c1.e[r] + bv)*scl;
    if (transp){
      outb[((size_t)nc0        << 10) + o] = (_Float16)v0;
      outb[((size_t)(nc0 + 16) << 10) + o] = (_Float16)v1;
    } else {
      outb[((size_t)o << 10) + nc0]      = (_Float16)v0;
      outb[((size_t)o << 10) + nc0 + 16] = (_Float16)v1;
    }
  }
}

// ---- Kernel 5: flash attention, register-prefetched K/V tiles ---------------
__global__ void __launch_bounds__(256) attn_kernel(
    const _Float16* __restrict__ projT, _Float16* __restrict__ attnT)
{
  __shared__ __align__(16) _Float16 Qs[64*LDQ];   // [q][d]
  __shared__ __align__(16) _Float16 Ks[64*LDQ];   // [t][d]
  __shared__ __align__(16) _Float16 Vs[128*LDT];  // [d][t]
  __shared__ __align__(16) _Float16 Ps[64*LDT];   // [q][t]
  __shared__ float Ss[64*LDS_S];
  __shared__ float m_run[64], l_run[64], alphaS[64], mnewS[64], part[256];

  const int tid = threadIdx.x, lane = tid & 31, wv = tid >> 5;
  const int qc = blockIdx.x, head = blockIdx.y;
  const int att = blockIdx.z >> 1, b = blockIdx.z & 1;
  const int qp = att ? 3 : 0, kp = att ? 4 : 1, vp = att ? 5 : 2;
  const _Float16* qT = projT + ((size_t)(qp*Bn + b) << 20);  // [s][c]
  const _Float16* kT = projT + ((size_t)(kp*Bn + b) << 20);  // [s][c]
  const _Float16* vv = projT + ((size_t)(vp*Bn + b) << 20);  // [c][s]
  const int q0 = qc*64, hc = head*HDIM;

  for (int j = tid; j < 1024; j += 256){
    int r2 = j >> 4, chk = (j & 15)*8;
    cp8(qT + (((size_t)(q0 + r2)) << 10) + hc + chk, Qs + r2*LDQ + chk);
  }
  if (tid < 64){ m_run[tid] = -1e30f; l_run[tid] = 0.f; }

  int kDst[4], vDst[4]; size_t kSrc[4], vSrc[4];
  #pragma unroll
  for (int i = 0; i < 4; ++i){
    int j = tid + i*256;
    kDst[i] = (j >> 4)*LDQ + (j & 15)*8;
    kSrc[i] = ((size_t)(j >> 4) << 10) + hc + (j & 15)*8;
    vDst[i] = (j >> 3)*LDT + (j & 7)*8;
    vSrc[i] = ((size_t)(hc + (j >> 3)) << 10) + (j & 7)*8;
  }
  uint4 rk0, rk1, rk2, rk3, rv0, rv1, rv2, rv3;
  auto fetchKV = [&](int t0){
    rk0 = *(const uint4*)(kT + ((size_t)t0 << 10) + kSrc[0]);
    rk1 = *(const uint4*)(kT + ((size_t)t0 << 10) + kSrc[1]);
    rk2 = *(const uint4*)(kT + ((size_t)t0 << 10) + kSrc[2]);
    rk3 = *(const uint4*)(kT + ((size_t)t0 << 10) + kSrc[3]);
    rv0 = *(const uint4*)(vv + vSrc[0] + t0);
    rv1 = *(const uint4*)(vv + vSrc[1] + t0);
    rv2 = *(const uint4*)(vv + vSrc[2] + t0);
    rv3 = *(const uint4*)(vv + vSrc[3] + t0);
  };
  fetchKV(0);

  Cf acc[4];
  #pragma unroll
  for (int t = 0; t < 4; ++t)
    #pragma unroll
    for (int i = 0; i < 8; ++i) acc[t].e[i] = 0.f;
  const int md0 = 16*wv;
  const int sxrow = tid & 63, sxq = (tid >> 6)*16;
  __syncthreads();

  for (int kt = 0; kt < 16; ++kt){
    *(uint4*)(Ks + kDst[0]) = rk0;  *(uint4*)(Ks + kDst[1]) = rk1;
    *(uint4*)(Ks + kDst[2]) = rk2;  *(uint4*)(Ks + kDst[3]) = rk3;
    *(uint4*)(Vs + vDst[0]) = rv0;  *(uint4*)(Vs + vDst[1]) = rv1;
    *(uint4*)(Vs + vDst[2]) = rv2;  *(uint4*)(Vs + vDst[3]) = rv3;
    __syncthreads();
    if (kt + 1 < 16) fetchKV((kt + 1)*64);
    { // S = Q^T K
      const int sm0 = 16*(wv & 3), sn0 = 32*(wv >> 2);
      Cf s0, s1;
      #pragma unroll
      for (int i = 0; i < 8; ++i){ s0.e[i] = 0.f; s1.e[i] = 0.f; }
      #pragma unroll
      for (int kk = 0; kk < 128; kk += 32){
        v16h a  = frag_a (Qs, LDQ, sm0, kk, lane);
        v16h b0 = frag_bk(Ks, LDQ, kk, sn0, lane);
        s0.v = WMMA_F32_F16(a, b0, s0.v);
        v16h b1 = frag_bk(Ks, LDQ, kk, sn0 + 16, lane);
        s1.v = WMMA_F32_F16(a, b1, s1.v);
      }
      const int g = (lane >> 4) & 1;
      const int col = sn0 + (lane & 15);
      #pragma unroll
      for (int r = 0; r < 8; ++r){
        int rw = sm0 + r + 8*g;
        Ss[rw*LDS_S + col]      = s0.e[r];
        Ss[rw*LDS_S + col + 16] = s1.e[r];
      }
    }
    __syncthreads();
    { // online softmax, 4 threads/row
      const float* srow = Ss + sxrow*LDS_S + sxq;
      float mt = -1e30f;
      #pragma unroll
      for (int j = 0; j < 16; ++j) mt = fmaxf(mt, srow[j]);
      part[tid] = mt;
      __syncthreads();
      if (tid < 64){
        float m4 = fmaxf(fmaxf(part[tid], part[64+tid]),
                         fmaxf(part[128+tid], part[192+tid]));
        float mnew = fmaxf(m_run[tid], m4);
        alphaS[tid] = __expf(m_run[tid] - mnew);
        mnewS[tid]  = mnew;
        m_run[tid]  = mnew;
      }
      __syncthreads();
      float mnew = mnewS[sxrow];
      _Float16* prow = Ps + sxrow*LDT + sxq;
      float sum = 0.f;
      #pragma unroll
      for (int j = 0; j < 16; ++j){
        float pv = __expf(srow[j] - mnew);
        prow[j] = (_Float16)pv;
        sum += pv;
      }
      part[tid] = sum;
      __syncthreads();
      if (tid < 64)
        l_run[tid] = l_run[tid]*alphaS[tid]
                   + part[tid] + part[64+tid] + part[128+tid] + part[192+tid];
      __syncthreads();
    }
    { // rescale + O += V * P^T
      #pragma unroll
      for (int ct = 0; ct < 4; ++ct){
        float al = alphaS[ct*16 + (lane & 15)];
        #pragma unroll
        for (int i = 0; i < 8; ++i) acc[ct].e[i] *= al;
      }
      #pragma unroll
      for (int kk = 0; kk < 64; kk += 32){
        v16h a = frag_a(Vs, LDT, md0, kk, lane);
        #pragma unroll
        for (int ct = 0; ct < 4; ++ct){
          v16h bt = frag_bk(Ps, LDT, kk, ct*16, lane);
          acc[ct].v = WMMA_F32_F16(a, bt, acc[ct].v);
        }
      }
    }
    __syncthreads();
  }
  _Float16* outb = attnT + ((size_t)(att*Bn + b) << 20);
  const int g = (lane >> 4) & 1;
  #pragma unroll
  for (int ct = 0; ct < 4; ++ct){
    int q = q0 + ct*16 + (lane & 15);
    float inv = 1.f / l_run[ct*16 + (lane & 15)];
    #pragma unroll
    for (int r = 0; r < 8; ++r){
      int d = md0 + r + 8*g;
      outb[((size_t)q << 10) + hc + d] = (_Float16)(acc[ct].e[r] * inv);
    }
  }
}

// ---- Kernel 6: fusion projection, K_TILE=64, register-prefetch --------------
__global__ void __launch_bounds__(256) fus_gemm_kernel(
    const _Float16* __restrict__ attnT,
    const float* __restrict__ fw, const float* __restrict__ fb,
    float* __restrict__ fusbuf)
{
  __shared__ __align__(16) _Float16 As [64*LDT];
  __shared__ __align__(16) _Float16 Bst[64*LDT];
  const int tid = threadIdx.x, lane = tid & 31, wv = tid >> 5;
  const int ntile = blockIdx.x, mtile = blockIdx.y, b = blockIdx.z;
  const int m0g = mtile*64, n0g = ntile*64;
  const int m0 = 16*(wv & 3), n0w = 32*(wv >> 2);
  const int row = tid >> 2;
  const int ch0 = (tid & 3)*8, ch1 = ch0 + 32;
  const int s2 = n0g + row;
  const int w  = s2 & 15;
  const int hd = s2 >> 4;
  // offset-select on a single base pointer (keeps loads on the GLOBAL path)
  const size_t boff = (w < 8)
      ? ((size_t)b        << 20) + (((size_t)(hd*8 + w))        << 10)
      : ((size_t)(Bn + b) << 20) + (((size_t)(hd*8 + (15 - w))) << 10);
  const _Float16* brow = attnT + boff;
  const float* wrow = fw + (size_t)(m0g + row)*Cch;
  Cf c0, c1;
  #pragma unroll
  for (int i = 0; i < 8; ++i){ c0.e[i] = 0.f; c1.e[i] = 0.f; }

  float4 wa0, wa1, wa2, wa3; uint4 rb0, rb1;
  auto fetch = [&](int k0){
    wa0 = *(const float4*)(wrow + k0 + ch0);
    wa1 = *(const float4*)(wrow + k0 + ch0 + 4);
    wa2 = *(const float4*)(wrow + k0 + ch1);
    wa3 = *(const float4*)(wrow + k0 + ch1 + 4);
    rb0 = *(const uint4*)(brow + k0 + ch0);
    rb1 = *(const uint4*)(brow + k0 + ch1);
  };
  fetch(0);
  for (int it = 0; it < 16; ++it){
    cvt8r(wa0, wa1, As + row*LDT + ch0);
    cvt8r(wa2, wa3, As + row*LDT + ch1);
    *(uint4*)(Bst + row*LDT + ch0) = rb0;
    *(uint4*)(Bst + row*LDT + ch1) = rb1;
    __syncthreads();
    if (it + 1 < 16) fetch((it + 1)*64);
    {
      v16h a0 = frag_a(As, LDT, m0, 0, lane);
      v16h bb;
      bb = frag_bk(Bst, LDT, 0, n0w,      lane); c0.v = WMMA_F32_F16(a0, bb, c0.v);
      bb = frag_bk(Bst, LDT, 0, n0w + 16, lane); c1.v = WMMA_F32_F16(a0, bb, c1.v);
    }
    {
      v16h a1 = frag_a(As, LDT, m0, 32, lane);
      v16h bb;
      bb = frag_bk(Bst, LDT, 32, n0w,      lane); c0.v = WMMA_F32_F16(a1, bb, c0.v);
      bb = frag_bk(Bst, LDT, 32, n0w + 16, lane); c1.v = WMMA_F32_F16(a1, bb, c1.v);
    }
    __syncthreads();
  }
  const int g = (lane >> 4) & 1;
  const int nc0 = n0g + n0w + (lane & 15);
  float* outb = fusbuf + (size_t)b*Cch*Sfull;
  #pragma unroll
  for (int r = 0; r < 8; ++r){
    int o = m0g + m0 + r + 8*g;
    float bv = fb[o];
    outb[(size_t)o*Sfull + nc0]      = c0.e[r] + bv;
    outb[(size_t)o*Sfull + nc0 + 16] = c1.e[r] + bv;
  }
}

// ---- Kernel 7: fusion GroupNorm + ReLU -> d_out -----------------------------
__global__ void __launch_bounds__(256) gn_fuse_kernel(
    const float* __restrict__ fusbuf,
    const float* __restrict__ gg, const float* __restrict__ gb,
    float* __restrict__ out)
{
  __shared__ float red[256], red2[256];
  __shared__ float s_mu, s_rs;
  const int tid = threadIdx.x;
  const int grp = blockIdx.x & 31, b = blockIdx.x >> 5;
  const float* src = fusbuf + (size_t)(b*Cch + grp*GCH)*Sfull;
  const int N = GCH * Sfull;
  float s = 0.f, ss = 0.f;
  for (int i = tid; i < N; i += 256){ float v = src[i]; s += v; ss += v*v; }
  red[tid] = s; red2[tid] = ss;
  __syncthreads();
  for (int off = 128; off > 0; off >>= 1){
    if (tid < off){ red[tid] += red[tid+off]; red2[tid] += red2[tid+off]; }
    __syncthreads();
  }
  if (tid == 0){
    float mu = red[0]/(float)N;
    s_mu = mu;
    s_rs = rsqrtf(red2[0]/(float)N - mu*mu + EPS);
  }
  __syncthreads();
  const float mu = s_mu, rs = s_rs;
  float* dst = out + (size_t)(b*Cch + grp*GCH)*Sfull;
  for (int i = tid; i < N; i += 256){
    int cc = i >> 11;
    int ch = grp*GCH + cc;
    float v = (src[i] - mu)*rs*gg[ch] + gb[ch];
    dst[i] = fmaxf(v, 0.f);
  }
}

// ---- launcher ---------------------------------------------------------------
extern "C" void kernel_launch(void* const* d_in, const int* in_sizes, int n_in,
                              void* d_out, int out_size, void* d_ws, size_t ws_size,
                              hipStream_t stream)
{
  (void)in_sizes; (void)n_in; (void)out_size; (void)ws_size;
  const float* x   = (const float*)d_in[0];
  const float* clw = (const float*)d_in[1];  const float* clb = (const float*)d_in[2];
  const float* glg = (const float*)d_in[3];  const float* glb = (const float*)d_in[4];
  const float* crw = (const float*)d_in[5];  const float* crb = (const float*)d_in[6];
  const float* grg = (const float*)d_in[7];  const float* grb = (const float*)d_in[8];
  const float* qlw = (const float*)d_in[9];  const float* qlb = (const float*)d_in[10];
  const float* krw = (const float*)d_in[11]; const float* krb = (const float*)d_in[12];
  const float* vrw = (const float*)d_in[13]; const float* vrb = (const float*)d_in[14];
  const float* qrw = (const float*)d_in[15]; const float* qrb = (const float*)d_in[16];
  const float* klw = (const float*)d_in[17]; const float* klb = (const float*)d_in[18];
  const float* vlw = (const float*)d_in[19]; const float* vlb = (const float*)d_in[20];
  const float* fw  = (const float*)d_in[21]; const float* fb  = (const float*)d_in[22];
  const float* gfg = (const float*)d_in[23]; const float* gfb = (const float*)d_in[24];
  float* out = (float*)d_out;

  // workspace (float units), ~177 MB total, with region reuse:
  //  [0,4M)    convbuf f32  -> reused as attnT f16
  //  [4M,8M)   fusbuf  f32  -> hosts xT f16 early
  //  [8M,10M)  gnT f16 ; [10M,16M) projT f16 ; [16M,43M) wT f16
  float* ws = (float*)d_ws;
  float*    convbuf = ws;
  _Float16* attnT   = (_Float16*)convbuf;
  float*    fusbuf  = ws + (size_t)4*1024*1024;
  _Float16* xT      = (_Float16*)fusbuf;
  _Float16* gnT     = (_Float16*)(ws + (size_t)8*1024*1024);
  _Float16* projT   = (_Float16*)(ws + (size_t)10*1024*1024);
  _Float16* wT      = (_Float16*)(ws + (size_t)16*1024*1024);

  dim3 blk(256);
  prep_x_kernel  <<<dim3(16384),     blk, 0, stream>>>(x, xT);
  prep_w_kernel  <<<dim3(4096,27,2), blk, 0, stream>>>(clw, crw, wT);
  conv3d_kernel  <<<dim3(16,16,4),   blk, 0, stream>>>(xT, wT, clb, crb, convbuf);
  gn_side_kernel <<<dim3(128),       blk, 0, stream>>>(convbuf, glg, glb, grg, grb, gnT);
  asym_kernel    <<<dim3(8),         blk, 0, stream>>>(gnT, out + 2*1024*2048);
  proj_kernel    <<<dim3(16,16,12),  blk, 0, stream>>>(gnT, qlw,qlb, krw,krb, vrw,vrb,
                                                       qrw,qrb, klw,klb, vlw,vlb, projT);
  attn_kernel    <<<dim3(16,8,4),    blk, 0, stream>>>(projT, attnT);
  fus_gemm_kernel<<<dim3(32,16,2),   blk, 0, stream>>>(attnT, fw, fb, fusbuf);
  gn_fuse_kernel <<<dim3(64),        blk, 0, stream>>>(fusbuf, gfg, gfb, out);
}
